// CausalSelfAttentionLayer_37830071943383
// MI455X (gfx1250) — matrix operand order repaired and checked by reading, not compiled, not run on hardware
//
#include <hip/hip_runtime.h>

typedef __attribute__((ext_vector_type(16))) _Float16 v16h;
typedef __attribute__((ext_vector_type(8)))  _Float16 v8h;
typedef __attribute__((ext_vector_type(8)))  float    v8f;

#define BATCH   4
#define SEQ     2048
#define EMB     1024
#define THREE_E 3072
#define NHEADS  16
#define HDIM    64

// ---------------------------------------------------------------------------
// fp32 -> f16 conversion
// ---------------------------------------------------------------------------
__global__ void cvt_f32_to_f16(const float* __restrict__ in,
                               _Float16* __restrict__ out, int n) {
    int i = blockIdx.x * blockDim.x + threadIdx.x;
    if (i < n) out[i] = (_Float16)in[i];
}

// ---------------------------------------------------------------------------
// WMMA fragment helpers (contiguous k order used by this code, wave32)
// A-matrix 16x32 (MxK): lanes 0-15 row M=lane hold K {0..15};
//                       lanes 16-31 row M=lane-16 hold K {16..31}
// ---------------------------------------------------------------------------
__device__ __forceinline__ v16h load_a_frag(const _Float16* __restrict__ A,
                                            int lda, int m0, int k0, int lane) {
    int row = m0 + (lane & 15);
    int kb  = k0 + ((lane >> 4) << 4);              // +16 for upper half-wave
    const _Float16* p = A + (size_t)row * lda + kb;
    v8h lo = *(const v8h*)p;                        // K = kb .. kb+7
    v8h hi = *(const v8h*)(p + 8);                  // K = kb+8 .. kb+15
    v16h a;
#pragma unroll
    for (int t = 0; t < 8; ++t) { a[t] = lo[t]; a[8 + t] = hi[t]; }
    return a;
}

// B-matrix 32x16 (KxN) built from row-major W[N][K] (i.e. B = W^T):
// lanes 0-15 col N=lane hold K {k0..k0+15}; lanes 16-31 hold K {k0+16..k0+31}
__device__ __forceinline__ v16h load_bT_frag(const _Float16* __restrict__ W,
                                             int ldw, int n0, int k0, int lane) {
    int col = n0 + (lane & 15);
    int kb  = k0 + ((lane >> 4) << 4);              // +16 for upper half-wave
    return *(const v16h*)(W + (size_t)col * ldw + kb);
}

__device__ __forceinline__ v8f wmma_f16(v16h a, v16h b, v8f c) {
    return __builtin_amdgcn_wmma_f32_16x16x32_f16(false, a, false, b,
                                                  (short)0, c, false, false);
}

// Async DMA: 2 x 16B  global -> LDS (ASYNCcnt-tracked, no VGPR bounce)
__device__ __forceinline__ void async_copy32(void* lds_dst, const void* gsrc) {
    unsigned l = (unsigned)(size_t)lds_dst;         // LDS byte offset (addr[31:0])
    asm volatile(
        "global_load_async_to_lds_b128 %0, %1, off\n\t"
        "global_load_async_to_lds_b128 %0, %1, off offset:16"
        :: "v"(l), "v"(gsrc) : "memory");
}

// Async DMA: 4 x 16B at 32B stride (64B of one V row) global -> LDS
__device__ __forceinline__ void async_copy128(void* lds_dst, const void* gsrc) {
    unsigned l = (unsigned)(size_t)lds_dst;
    asm volatile(
        "global_load_async_to_lds_b128 %0, %1, off\n\t"
        "global_load_async_to_lds_b128 %0, %1, off offset:32\n\t"
        "global_load_async_to_lds_b128 %0, %1, off offset:64\n\t"
        "global_load_async_to_lds_b128 %0, %1, off offset:96"
        :: "v"(l), "v"(gsrc) : "memory");
}

__device__ __forceinline__ void wait_async0() {
    asm volatile("s_wait_asynccnt 0" ::: "memory");
}

// ---------------------------------------------------------------------------
// QKV projection: block = 64(M) x 64(N) tile; 4 waves, B panel shared via LDS
// (staged with async-to-LDS DMA). Epilogue adds bias and scatters to q/k/v in
// [B,H,S,D] f16 layout.
// ---------------------------------------------------------------------------
__global__ void gemm_qkv_wmma(const _Float16* __restrict__ x16,
                              const _Float16* __restrict__ w16,
                              const float* __restrict__ bias,
                              _Float16* __restrict__ q16,
                              _Float16* __restrict__ k16,
                              _Float16* __restrict__ v16) {
    __shared__ __align__(32) _Float16 bpanel[64][32];

    const int lane = threadIdx.x & 31;
    const int wib  = threadIdx.x >> 5;
    const int nbN  = THREE_E / 64;                  // 48
    const int m0 = (blockIdx.x / nbN) * 64 + wib * 16;
    const int n0 = (blockIdx.x % nbN) * 64;

    v8f acc[4];
#pragma unroll
    for (int c = 0; c < 4; ++c) acc[c] = (v8f)0.f;

    const int pcol = threadIdx.x >> 1, pch = threadIdx.x & 1;

    for (int k0 = 0; k0 < EMB; k0 += 32) {
        // stage shared B panel (64 cols x 32 k) via async DMA
        async_copy32(&bpanel[pcol][pch * 16],
                     w16 + (size_t)(n0 + pcol) * EMB + k0 + pch * 16);
        __builtin_prefetch(x16 + (size_t)m0 * EMB + k0 + 32, 0, 1);
        __builtin_prefetch(w16 + (size_t)n0 * EMB + k0 + 32, 0, 1);
        wait_async0();
        __syncthreads();

        v16h a = load_a_frag(x16, EMB, m0, k0, lane);
#pragma unroll
        for (int c = 0; c < 4; ++c) {
            v16h b = *(const v16h*)&bpanel[16 * c + (lane & 15)][(lane >> 4) * 16];
            acc[c] = wmma_f16(a, b, acc[c]);
        }
        __syncthreads();
    }

    const int grp  = lane >> 4;
    const int colL = lane & 15;
#pragma unroll
    for (int c = 0; c < 4; ++c) {
#pragma unroll
        for (int i = 0; i < 8; ++i) {
            int n = n0 + 16 * c + colL;
            int m = m0 + i + 8 * grp;
            float val = acc[c][i] + bias[n];
            int which = n >> 10;                    // 0:q 1:k 2:v
            int e  = n & (EMB - 1);
            int h  = e >> 6;
            int dd = e & (HDIM - 1);
            int s  = m & (SEQ - 1);
            int bb = m >> 11;
            _Float16* dst = (which == 0) ? q16 : ((which == 1) ? k16 : v16);
            dst[(((size_t)(bb * NHEADS + h) * SEQ + s)) * HDIM + dd] = (_Float16)val;
        }
    }
}

// ---------------------------------------------------------------------------
// Flash attention: one wave per (b,h, 16-row q tile); 32-key blocks.
// Full K=32 PV WMMA; V tile DMA'd to LDS asynchronously, overlapping QK^T.
// ---------------------------------------------------------------------------
__global__ void flash_attn_wmma(const _Float16* __restrict__ q16,
                                const _Float16* __restrict__ k16,
                                const _Float16* __restrict__ v16,
                                _Float16* __restrict__ y16) {
    __shared__ __align__(32) _Float16 ptile[4][16][32];
    __shared__ __align__(32) _Float16 vtile[4][32][64];

    const int lane = threadIdx.x & 31;
    const int wib  = threadIdx.x >> 5;
    const int wave = blockIdx.x * (blockDim.x >> 5) + wib;
    const int qtiles = SEQ / 16;                    // 128
    const int qt = wave % qtiles;
    const int bh = wave / qtiles;                   // 0..63
    const int q0 = qt * 16;
    const int grp  = lane >> 4;
    const int colL = lane & 15;

    const _Float16* Q = q16 + (size_t)bh * SEQ * HDIM;
    const _Float16* K = k16 + (size_t)bh * SEQ * HDIM;
    const _Float16* V = v16 + (size_t)bh * SEQ * HDIM;

    // Q fragments for full D=64 (two K=32 slabs), loaded once
    v16h qa0 = load_a_frag(Q, HDIM, q0, 0, lane);
    v16h qa1 = load_a_frag(Q, HDIM, q0, 32, lane);

    v8f yacc[4];
#pragma unroll
    for (int c = 0; c < 4; ++c) yacc[c] = (v8f)0.f;
    float mrow[8], lrow[8];
#pragma unroll
    for (int i = 0; i < 8; ++i) { mrow[i] = -3.0e38f; lrow[i] = 0.f; }

    const int jmax = q0 >> 5;                       // last 32-key block index
    for (int j = 0; j <= jmax; ++j) {
        const int kb = j * 32;

        // ---- kick off async V-tile DMA first: overlaps with QK^T WMMAs ----
        // (lane r copies 4 x 16B of V row kb+r, 32B apart, straight into LDS)
        async_copy128(&vtile[wib][lane][0], V + (size_t)(kb + lane) * HDIM);
        __builtin_prefetch(K + (size_t)(kb + 32) * HDIM, 0, 1);
        __builtin_prefetch(V + (size_t)(kb + 32) * HDIM, 0, 1);

        // ---- S = Q K^T : two 16-col score tiles, K=64 in two 32-slabs -----
        v16h kA0 = load_bT_frag(K, HDIM, kb,       0, lane);
        v16h kA1 = load_bT_frag(K, HDIM, kb,      32, lane);
        v16h kB0 = load_bT_frag(K, HDIM, kb + 16,  0, lane);
        v16h kB1 = load_bT_frag(K, HDIM, kb + 16, 32, lane);
        v8f s0 = (v8f)0.f, s1 = (v8f)0.f;
        s0 = wmma_f16(qa0, kA0, s0);
        s0 = wmma_f16(qa1, kA1, s0);
        s1 = wmma_f16(qa0, kB0, s1);
        s1 = wmma_f16(qa1, kB1, s1);

        // ---- scale + causal mask (only last block can touch the diagonal) -
        const bool last = (j == jmax);
#pragma unroll
        for (int i = 0; i < 8; ++i) {
            float a0 = s0[i] * 0.125f;              // 1/sqrt(64)
            float a1 = s1[i] * 0.125f;
            if (last) {
                int r = q0 + i + 8 * grp;
                if (kb + colL > r)      a0 = -1.0e30f;
                if (kb + 16 + colL > r) a1 = -1.0e30f;
            }
            s0[i] = a0; s1[i] = a1;
        }

        // ---- online softmax: one reduction chain per row per 32 keys ------
#pragma unroll
        for (int i = 0; i < 8; ++i) {
            float v = fmaxf(s0[i], s1[i]);
#pragma unroll
            for (int off = 1; off < 16; off <<= 1)
                v = fmaxf(v, __shfl_xor(v, off, 16));
            float mnew  = fmaxf(mrow[i], v);
            float alpha = __expf(mrow[i] - mnew);
            float p0    = __expf(s0[i] - mnew);
            float p1    = __expf(s1[i] - mnew);
            float psum  = p0 + p1;
#pragma unroll
            for (int off = 1; off < 16; off <<= 1)
                psum += __shfl_xor(psum, off, 16);
            lrow[i] = lrow[i] * alpha + psum;
            mrow[i] = mnew;
#pragma unroll
            for (int c = 0; c < 4; ++c) yacc[c][i] *= alpha;
            int row = i + 8 * grp;
            ptile[wib][row][colL]      = (_Float16)p0;
            ptile[wib][row][16 + colL] = (_Float16)p1;
        }

        // V DMA + P stores must land before re-striping from LDS
        asm volatile("s_wait_asynccnt 0\n\t"
                     "s_wait_dscnt 0" ::: "memory");

        // ---- P as a full 16x32 A-fragment ---------------------------------
        v16h pa;
        {
            const _Float16* pr = &ptile[wib][colL][grp * 16];
            v8h lo = *(const v8h*)pr;               // K grp*16 .. +7
            v8h hi = *(const v8h*)(pr + 8);         // K grp*16+8 .. +15
#pragma unroll
            for (int t = 0; t < 8; ++t) { pa[t] = lo[t]; pa[8 + t] = hi[t]; }
        }

        // ---- Y += P * V over four 16-wide d-chunks (full K=32) ------------
#pragma unroll
        for (int c = 0; c < 4; ++c) {
            v16h vb;
            int dcol = c * 16 + colL;
            int kofs = grp * 16;
#pragma unroll
            for (int h = 0; h < 16; ++h)
                vb[h] = vtile[wib][kofs + h][dcol];
            yacc[c] = wmma_f16(pa, vb, yacc[c]);
        }
    }

    // normalize + store to y16 in [B,S,E] layout (fuses head transpose)
    const int bb = bh / NHEADS;
    const int h  = bh % NHEADS;
#pragma unroll
    for (int c = 0; c < 4; ++c) {
#pragma unroll
        for (int i = 0; i < 8; ++i) {
            int srow = q0 + i + 8 * grp;
            float val = yacc[c][i] / lrow[i];
            y16[((size_t)(bb * SEQ + srow)) * EMB + h * HDIM + c * 16 + colL] =
                (_Float16)val;
        }
    }
}

// ---------------------------------------------------------------------------
// Output projection: block = 64x64 tile, async-staged LDS B panel, fp32 out.
// ---------------------------------------------------------------------------
__global__ void gemm_proj_wmma(const _Float16* __restrict__ y16,
                               const _Float16* __restrict__ w16,
                               const float* __restrict__ bias,
                               float* __restrict__ out) {
    __shared__ __align__(32) _Float16 bpanel[64][32];

    const int lane = threadIdx.x & 31;
    const int wib  = threadIdx.x >> 5;
    const int nbN  = EMB / 64;                      // 16
    const int m0 = (blockIdx.x / nbN) * 64 + wib * 16;
    const int n0 = (blockIdx.x % nbN) * 64;

    v8f acc[4];
#pragma unroll
    for (int c = 0; c < 4; ++c) acc[c] = (v8f)0.f;

    const int pcol = threadIdx.x >> 1, pch = threadIdx.x & 1;

    for (int k0 = 0; k0 < EMB; k0 += 32) {
        async_copy32(&bpanel[pcol][pch * 16],
                     w16 + (size_t)(n0 + pcol) * EMB + k0 + pch * 16);
        __builtin_prefetch(y16 + (size_t)m0 * EMB + k0 + 32, 0, 1);
        __builtin_prefetch(w16 + (size_t)n0 * EMB + k0 + 32, 0, 1);
        wait_async0();
        __syncthreads();

        v16h a = load_a_frag(y16, EMB, m0, k0, lane);
#pragma unroll
        for (int c = 0; c < 4; ++c) {
            v16h b = *(const v16h*)&bpanel[16 * c + (lane & 15)][(lane >> 4) * 16];
            acc[c] = wmma_f16(a, b, acc[c]);
        }
        __syncthreads();
    }

    const int grp  = lane >> 4;
    const int colL = lane & 15;
#pragma unroll
    for (int c = 0; c < 4; ++c) {
#pragma unroll
        for (int i = 0; i < 8; ++i) {
            int n = n0 + 16 * c + colL;
            int m = m0 + i + 8 * grp;
            out[(size_t)m * EMB + n] = acc[c][i] + bias[n];
        }
    }
}

// ---------------------------------------------------------------------------
extern "C" void kernel_launch(void* const* d_in, const int* in_sizes, int n_in,
                              void* d_out, int out_size, void* d_ws, size_t ws_size,
                              hipStream_t stream) {
    const float* x      = (const float*)d_in[0];
    const float* W_kqv  = (const float*)d_in[1];
    const float* b_kqv  = (const float*)d_in[2];
    const float* W_proj = (const float*)d_in[3];
    const float* b_proj = (const float*)d_in[4];
    float* out = (float*)d_out;

    _Float16* ws = (_Float16*)d_ws;
    size_t o = 0;
    _Float16* x16  = ws + o; o += (size_t)BATCH * SEQ * EMB;
    _Float16* wk16 = ws + o; o += (size_t)THREE_E * EMB;
    _Float16* wp16 = ws + o; o += (size_t)EMB * EMB;
    _Float16* q16  = ws + o; o += (size_t)BATCH * NHEADS * SEQ * HDIM;
    _Float16* k16  = ws + o; o += (size_t)BATCH * NHEADS * SEQ * HDIM;
    _Float16* v16  = ws + o; o += (size_t)BATCH * NHEADS * SEQ * HDIM;
    _Float16* y16  = ws + o; o += (size_t)BATCH * SEQ * EMB;
    (void)ws_size; (void)in_sizes; (void)n_in; (void)out_size;

    const int nx = BATCH * SEQ * EMB;
    cvt_f32_to_f16<<<(nx + 255) / 256, 256, 0, stream>>>(x, x16, nx);
    const int nwk = THREE_E * EMB;
    cvt_f32_to_f16<<<(nwk + 255) / 256, 256, 0, stream>>>(W_kqv, wk16, nwk);
    const int nwp = EMB * EMB;
    cvt_f32_to_f16<<<(nwp + 255) / 256, 256, 0, stream>>>(W_proj, wp16, nwp);

    // QKV: (B*S/64) * (3E/64) = 128 * 48 = 6144 blocks of 128 threads
    gemm_qkv_wmma<<<6144, 128, 0, stream>>>(x16, wk16, b_kqv, q16, k16, v16);

    // Attention: B*H*(S/16) = 8192 waves, 4 waves/block
    flash_attn_wmma<<<8192 / 4, 128, 0, stream>>>(q16, k16, v16, y16);

    // Proj: (B*S/64) * (E/64) = 128 * 16 = 2048 blocks
    gemm_proj_wmma<<<2048, 128, 0, stream>>>(y16, wp16, b_proj, out);
}
